// Transformer_77962246357182
// MI455X (gfx1250) — matrix operand order repaired and checked by reading, not compiled, not run on hardware
//
#include <hip/hip_runtime.h>

// ---------------------------------------------------------------------------
// CDNA5 / gfx1250 attention pipeline:
//   keys = search_x @ Wk        (WMMA bf16 GEMM -> bf16 ws)
//   qk   = query_x @ Wq         (WMMA bf16 GEMM -> bf16 ws)
//   out  = softmax_mask(qk keys^T * s) @ search_y   (fused flash kernel, WMMA)
// Key tiles are double-buffered in LDS and prefetched with
// GLOBAL_LOAD_ASYNC_TO_LDS_B128 (ASYNCcnt) so the 64KB copy overlaps the
// whole compute of the previous tile. bf16 pair-packing via v_perm_b32.
// ---------------------------------------------------------------------------

typedef __attribute__((ext_vector_type(16))) __bf16 v16bf;
typedef __attribute__((ext_vector_type(8)))  __bf16 v8bf;
typedef __attribute__((ext_vector_type(2)))  __bf16 v2bf;
typedef __attribute__((ext_vector_type(8)))  float  v8f;
typedef __attribute__((ext_vector_type(4)))  int    v4i;

static constexpr int kN  = 8192;
static constexpr int kD  = 1024;
static constexpr int kQ  = 4096;
static constexpr int kDK = 512;
static constexpr int kDV = 256;

#if defined(__has_builtin)
#if __has_builtin(__builtin_amdgcn_global_load_async_to_lds_b128)
#define HAVE_ASYNC_CP 1
#endif
#if __has_builtin(__builtin_amdgcn_cvt_pk_bf16_f32)
#define HAVE_CVT_PK 1
#endif
#if __has_builtin(__builtin_amdgcn_perm)
#define HAVE_PERM 1
#endif
#endif
#ifndef HAVE_ASYNC_CP
#define HAVE_ASYNC_CP 0
#endif
#ifndef HAVE_CVT_PK
#define HAVE_CVT_PK 0
#endif
#ifndef HAVE_PERM
#define HAVE_PERM 0
#endif

// two f32 -> packed bf16 pair (low = a, high = b), round-to-nearest
static __device__ __forceinline__ unsigned int f2bf2(float a, float b) {
#if HAVE_CVT_PK
  v2bf r = __builtin_amdgcn_cvt_pk_bf16_f32(a, b);
  return __builtin_bit_cast(unsigned int, r);
#else
  const unsigned int ua = __float_as_uint(a) + 0x8000u;
  const unsigned int ub = __float_as_uint(b) + 0x8000u;
#if HAVE_PERM
  // dst bytes {0,1} = ua bytes {2,3}; dst bytes {2,3} = ub bytes {2,3}
  return __builtin_amdgcn_perm(ub, ua, 0x07060302u);
#else
  return (ua >> 16) | (ub & 0xffff0000u);
#endif
#endif
}

// 16-byte global -> LDS copy; async (no VGPR roundtrip) when available.
static __device__ __forceinline__ void async_cp16(void* lds_dst, const void* gsrc) {
#if HAVE_ASYNC_CP
  void* s = const_cast<void*>(gsrc);
  __attribute__((address_space(1))) v4i* g =
      (__attribute__((address_space(1))) v4i*)s;
  __attribute__((address_space(3))) v4i* l =
      (__attribute__((address_space(3))) v4i*)lds_dst;
  __builtin_amdgcn_global_load_async_to_lds_b128(g, l, 0, 0);
#else
  *(uint4*)lds_dst = *(const uint4*)gsrc;
#endif
}
static __device__ __forceinline__ void wait_async0() {
#if HAVE_ASYNC_CP
#if defined(__has_builtin) && __has_builtin(__builtin_amdgcn_s_wait_asynccnt)
  __builtin_amdgcn_s_wait_asynccnt(0);
#else
  asm volatile("s_wait_asynccnt 0x0" ::: "memory");
#endif
#endif
}

static __device__ __forceinline__ v8f wmma_bf16(v16bf a, v16bf b, v8f c) {
  return __builtin_amdgcn_wmma_f32_16x16x32_bf16(false, a, false, b, (short)0, c,
                                                 false, false);
}

// A-fragment: two contiguous 8-element runs at base and base+16 (u16 units).
static __device__ __forceinline__ v16bf ld_frag_a(const unsigned short* base) {
  const v8bf lo = *(const v8bf*)(base);
  const v8bf hi = *(const v8bf*)(base + 16);
  return __builtin_shufflevector(lo, hi, 0, 1, 2, 3, 4, 5, 6, 7,
                                 8, 9, 10, 11, 12, 13, 14, 15);
}
// B-fragment from a transposed-row source: 16 contiguous bf16 at base.
static __device__ __forceinline__ v16bf ld_frag_row16(const unsigned short* base) {
  const v8bf lo = *(const v8bf*)(base);
  const v8bf hi = *(const v8bf*)(base + 8);
  return __builtin_shufflevector(lo, hi, 0, 1, 2, 3, 4, 5, 6, 7,
                                 8, 9, 10, 11, 12, 13, 14, 15);
}

union V16U { unsigned int u[8]; v16bf v; };

// ---------------------------------------------------------------------------
// Projection GEMM: C[M x 512] = X[M x 1024] @ W[1024 x 512], output bf16.
// Block: 256 threads (8 waves), tile 64(M) x 64(N), K steps of 32.
// W tile staged k-pair interleaved: (k,n) -> Bs2[k>>1][n*2 + (k&1)].
// ---------------------------------------------------------------------------
__global__ __launch_bounds__(256) void proj_gemm(const float* __restrict__ X,
                                                 const float* __restrict__ W,
                                                 unsigned short* __restrict__ Obf) {
  __shared__ unsigned short As[64][40];    // 64 x 32 bf16, padded
  __shared__ unsigned short Bs2[16][136];  // 16 k-pair rows x (64 cols *2 + pad)

  const int tid  = threadIdx.x;
  const int lane = tid & 31;
  const int wid  = tid >> 5;
  const int mr   = wid >> 1;      // 0..3 : 16-row band
  const int nh   = wid & 1;       // 0/1  : 32-col band
  const int m0   = blockIdx.x * 64;
  const int n0   = blockIdx.y * 64;

  v8f c0 = {}; v8f c1 = {};

  const int axr = tid >> 2;        // A staging: row 0..63
  const int axc = (tid & 3) * 8;
  const int bk_ = tid >> 3;        // B staging: k 0..31
  const int bg_ = (tid & 7) * 8;   //            col base

  const int ar  = mr * 16 + (lane & 15);
  const int kh8 = (lane >> 4) * 8;
  const int bc0 = nh * 32 + (lane & 15);
  const int rb  = (lane >> 4) * 8;  // k-pair row base for B frags

  for (int k0 = 0; k0 < kD; k0 += 32) {
    {
      const float4 v0 = *(const float4*)(X + (size_t)(m0 + axr) * kD + k0 + axc);
      const float4 v1 = *(const float4*)(X + (size_t)(m0 + axr) * kD + k0 + axc + 4);
      const uint4 q = make_uint4(f2bf2(v0.x, v0.y), f2bf2(v0.z, v0.w),
                                 f2bf2(v1.x, v1.y), f2bf2(v1.z, v1.w));
      *(uint4*)&As[axr][axc] = q;
    }
    {
      const float4 v0 = *(const float4*)(W + (size_t)(k0 + bk_) * kDK + n0 + bg_);
      const float4 v1 = *(const float4*)(W + (size_t)(k0 + bk_) * kDK + n0 + bg_ + 4);
      unsigned int pk[4];
      pk[0] = f2bf2(v0.x, v0.y); pk[1] = f2bf2(v0.z, v0.w);
      pk[2] = f2bf2(v1.x, v1.y); pk[3] = f2bf2(v1.z, v1.w);
      unsigned short* dst = &Bs2[bk_ >> 1][bk_ & 1];
#pragma unroll
      for (int j = 0; j < 4; ++j) {
        dst[(bg_ + 2 * j) * 2]     = (unsigned short)pk[j];
        dst[(bg_ + 2 * j + 1) * 2] = (unsigned short)(pk[j] >> 16);
      }
    }
    __syncthreads();

    const v16bf a = ld_frag_row16(&As[ar][kh8 * 2]);
    V16U b0, b1;
#pragma unroll
    for (int i = 0; i < 8; ++i) {
      b0.u[i] = *(const unsigned int*)&Bs2[rb + i][bc0 * 2];
      b1.u[i] = *(const unsigned int*)&Bs2[rb + i][(bc0 + 16) * 2];
    }
    c0 = wmma_bf16(a, b0.v, c0);
    c1 = wmma_bf16(a, b1.v, c1);
    __syncthreads();
  }

#pragma unroll
  for (int r = 0; r < 8; ++r) {
    const int row = mr * 16 + r + (lane >> 4) * 8;
    const int col = nh * 32 + (lane & 15);
    const unsigned int pk = f2bf2(c0[r], c1[r]);   // cols col and col+16
    Obf[(size_t)(m0 + row) * kDK + n0 + col]      = (unsigned short)pk;
    Obf[(size_t)(m0 + row) * kDK + n0 + col + 16] = (unsigned short)(pk >> 16);
  }
}

// ---------------------------------------------------------------------------
// f32 -> bf16 conversion (search_y)
// ---------------------------------------------------------------------------
__global__ __launch_bounds__(256) void cvt_bf16(const float* __restrict__ src,
                                                unsigned short* __restrict__ dst,
                                                int n4) {
  const int i = blockIdx.x * 256 + threadIdx.x;
  if (i < n4) {
    const float4 v = ((const float4*)src)[i];
    *(uint2*)(dst + (size_t)i * 4) = make_uint2(f2bf2(v.x, v.y), f2bf2(v.z, v.w));
  }
}

// ---------------------------------------------------------------------------
// Fused flash attention. 256 threads (8 waves), 32 query rows, key tiles 64.
//   mr = wid>>2 (query 16-row band), nq = wid&3.
//   S phase : one 16x16 WMMA tile per wave (cols nq*16..+16).
//   PV phase: rows mr-band x cols [nq*64, nq*64+64) -> four accumulators.
// Key tiles double-buffered; async prefetch of tile t+1 overlaps tile t math.
// LDS: Qs 32x520 | Ks0/Ks1 64x520 | Yt 32x520 (k-pair interleaved) |
//      Ss 32x68 f32 | Ps 32x72 | Ms 32x64 u8 | m/l/alpha
// ---------------------------------------------------------------------------
static constexpr int QSL = 520;
static constexpr int KSL = 520;
static constexpr int YTL = 520;   // 256 cols * 2 + 8 pad (u16 units)
static constexpr int SSL = 68;
static constexpr int PSL = 72;

static constexpr int OFF_QS  = 0;
static constexpr int OFF_KS0 = OFF_QS  + 32 * QSL * 2;  //  33280
static constexpr int OFF_KS1 = OFF_KS0 + 64 * KSL * 2;  //  99840
static constexpr int OFF_YT  = OFF_KS1 + 64 * KSL * 2;  // 166400
static constexpr int OFF_SS  = OFF_YT  + 32 * YTL * 2;  // 199680
static constexpr int OFF_PS  = OFF_SS  + 32 * SSL * 4;  // 208384
static constexpr int OFF_MS  = OFF_PS  + 32 * PSL * 2;  // 212992
static constexpr int OFF_RED = OFF_MS  + 32 * 64;       // 215040
static constexpr int SMEM_BYTES = OFF_RED + 3 * 32 * 4; // 215424

__global__ __launch_bounds__(256) void flash_attn(
    const unsigned short* __restrict__ qk,    // [4096][512] bf16
    const unsigned short* __restrict__ keys,  // [8192][512] bf16
    const unsigned short* __restrict__ yv,    // [8192][256] bf16
    const long long* __restrict__ mask,       // [4096][8192]
    float* __restrict__ out) {                // [4096][256]
  extern __shared__ char smem[];
  unsigned short* Qs  = (unsigned short*)(smem + OFF_QS);
  unsigned short* Ks0 = (unsigned short*)(smem + OFF_KS0);
  unsigned short* Ks1 = (unsigned short*)(smem + OFF_KS1);
  unsigned short* Yt  = (unsigned short*)(smem + OFF_YT);
  float*          Ss  = (float*)(smem + OFF_SS);
  unsigned short* Ps  = (unsigned short*)(smem + OFF_PS);
  unsigned char*  Ms  = (unsigned char*)(smem + OFF_MS);
  float* mRow = (float*)(smem + OFF_RED);
  float* lRow = mRow + 32;
  float* aRow = lRow + 32;

  const int tid  = threadIdx.x;
  const int lane = tid & 31;
  const int wid  = tid >> 5;
  const int mr   = wid >> 2;   // 0..1
  const int nq   = wid & 3;    // 0..3
  const int q0   = blockIdx.x * 32;

  // ---- startup: async-stage Qs and key tile 0 ----
  {
    const unsigned short* src = qk + (size_t)q0 * kDK;
    for (int idx = tid; idx < 32 * 64; idx += 256) {
      const int r = idx >> 6, c = idx & 63;
      async_cp16(Qs + (size_t)r * QSL + c * 8, src + (size_t)r * kDK + c * 8);
    }
  }
  for (int idx = tid; idx < 64 * 64; idx += 256) {
    const int r = idx >> 6, c = idx & 63;
    async_cp16(Ks0 + (size_t)r * KSL + c * 8, keys + (size_t)r * kDK + c * 8);
  }
  if (tid < 32) { mRow[tid] = -3.0e38f; lRow[tid] = 0.0f; }
  wait_async0();
  __syncthreads();

  v8f o[4] = {};
  const float scale = 0.044194173824159216f;  // 1/sqrt(512)

  const int arow = mr * 16 + (lane & 15);
  const int kh8  = (lane >> 4) * 8;
  const int kh16 = (lane >> 4) * 16;
  const int bcol = nq * 16 + (lane & 15);   // S-phase B column = key row
  const int pr   = mr * 16 + (lane & 15);

  for (int nt = 0; nt < kN / 64; ++nt) {
    const int n0 = nt * 64;
    unsigned short* Kcur = (nt & 1) ? Ks1 : Ks0;
    unsigned short* Knxt = (nt & 1) ? Ks0 : Ks1;

    // ---- stage value tile (k-pair interleaved) and mask tile (manual) ----
    for (int idx = tid; idx < 64 * 32; idx += 256) {   // (key row, 8-col group)
      const int k = idx >> 5;
      const int g = (idx & 31) * 8;
      const uint4 v = *(const uint4*)(yv + (size_t)(n0 + k) * kDV + g);
      const unsigned short* e = (const unsigned short*)&v;
      unsigned short* dst = Yt + (size_t)(k >> 1) * YTL + (k & 1);
#pragma unroll
      for (int j = 0; j < 8; ++j) dst[(g + j) * 2] = e[j];
    }
    for (int idx = tid; idx < 32 * 64; idx += 256) {
      const int r = idx >> 6, c = idx & 63;
      Ms[idx] = (mask[(size_t)(q0 + r) * kN + n0 + c] > 0) ? 1 : 0;
    }

    // ---- wait for this tile's async key copy, then prefetch next tile ----
    wait_async0();
    if (nt + 1 < kN / 64) {
      const unsigned short* src = keys + (size_t)(n0 + 64) * kDK;
      for (int idx = tid; idx < 64 * 64; idx += 256) {
        const int r = idx >> 6, c = idx & 63;
        async_cp16(Knxt + (size_t)r * KSL + c * 8, src + (size_t)r * kDK + c * 8);
      }
    }
    __syncthreads();   // publish Kcur, Yt, Ms

    // ---- S = qk @ keys^T (one 16x16 tile per wave) ----
    v8f s = {};
#pragma unroll
    for (int ks = 0; ks < 16; ++ks) {
      const v16bf a = ld_frag_row16(Qs + (size_t)arow * QSL + ks * 32 + kh16);
      const v16bf b = ld_frag_row16(Kcur + (size_t)bcol * KSL + ks * 32 + kh16);
      s = wmma_bf16(a, b, s);
    }
#pragma unroll
    for (int r = 0; r < 8; ++r) {
      const int row = mr * 16 + r + (lane >> 4) * 8;
      Ss[row * SSL + nq * 16 + (lane & 15)] = s[r];
    }
    __syncthreads();

    // ---- masked online softmax: one thread per query row ----
    if (tid < 32) {
      const float mOld = mRow[tid];
      float mNew = mOld;
      for (int j = 0; j < 64; ++j)
        if (Ms[tid * 64 + j]) mNew = fmaxf(mNew, Ss[tid * SSL + j] * scale);
      float sum = 0.0f;
      for (int j = 0; j < 64; j += 2) {
        float p0 = 0.0f, p1 = 0.0f;
        if (Ms[tid * 64 + j])     p0 = __expf(Ss[tid * SSL + j] * scale - mNew);
        if (Ms[tid * 64 + j + 1]) p1 = __expf(Ss[tid * SSL + j + 1] * scale - mNew);
        *(unsigned int*)&Ps[tid * PSL + j] = f2bf2(p0, p1);
        sum += p0 + p1;
      }
      const float alpha = (mNew > mOld) ? __expf(mOld - mNew) : 1.0f;
      lRow[tid] = lRow[tid] * alpha + sum;
      mRow[tid] = mNew;
      aRow[tid] = alpha;
    }
    __syncthreads();

    // ---- O = O*alpha + P @ Y ----
    float al[8];
#pragma unroll
    for (int r = 0; r < 8; ++r) al[r] = aRow[mr * 16 + r + (lane >> 4) * 8];
#pragma unroll
    for (int j = 0; j < 4; ++j)
#pragma unroll
      for (int r = 0; r < 8; ++r) o[j][r] *= al[r];

#pragma unroll
    for (int ks = 0; ks < 2; ++ks) {
      const v16bf ap = ld_frag_row16(Ps + (size_t)pr * PSL + ks * 32 + kh16);
      const int rbase = ks * 16 + (lane >> 4) * 8;  // k-pair row base
#pragma unroll
      for (int j = 0; j < 4; ++j) {
        const int c0 = nq * 64 + j * 16 + (lane & 15);
        V16U b;
#pragma unroll
        for (int i = 0; i < 8; ++i)
          b.u[i] = *(const unsigned int*)(Yt + (size_t)(rbase + i) * YTL + c0 * 2);
        o[j] = wmma_bf16(ap, b.v, o[j]);
      }
    }
    __syncthreads();
  }

  // ---- epilogue: O / l ----
#pragma unroll
  for (int j = 0; j < 4; ++j) {
#pragma unroll
    for (int r = 0; r < 8; ++r) {
      const int row = mr * 16 + r + (lane >> 4) * 8;
      const float l = lRow[row];
      const float inv = (l > 0.0f) ? 1.0f / l : 0.0f;
      out[(size_t)(q0 + row) * kDV + nq * 64 + j * 16 + (lane & 15)] = o[j][r] * inv;
    }
  }
}

// ---------------------------------------------------------------------------
extern "C" void kernel_launch(void* const* d_in, const int* in_sizes, int n_in,
                              void* d_out, int out_size, void* d_ws, size_t ws_size,
                              hipStream_t stream) {
  (void)in_sizes; (void)n_in; (void)out_size; (void)ws_size;
  const float*     search_x = (const float*)d_in[0];
  const float*     search_y = (const float*)d_in[1];
  const float*     query_x  = (const float*)d_in[2];
  const long long* mask     = (const long long*)d_in[3];
  const float*     Wk       = (const float*)d_in[4];
  const float*     Wq       = (const float*)d_in[5];
  float*           out      = (float*)d_out;

  unsigned short* keys_bf = (unsigned short*)d_ws;
  unsigned short* qk_bf   = (unsigned short*)((char*)d_ws + (size_t)kN * kDK * 2);
  unsigned short* y_bf    = (unsigned short*)((char*)d_ws + (size_t)kN * kDK * 2 +
                                              (size_t)kQ * kDK * 2);

  proj_gemm<<<dim3(kN / 64, kDK / 64), 256, 0, stream>>>(search_x, Wk, keys_bf);
  proj_gemm<<<dim3(kQ / 64, kDK / 64), 256, 0, stream>>>(query_x, Wq, qk_bf);
  cvt_bf16<<<(kN * kDV / 4 + 255) / 256, 256, 0, stream>>>(search_y, y_bf,
                                                           kN * kDV / 4);

  (void)hipFuncSetAttribute((const void*)flash_attn,
                            hipFuncAttributeMaxDynamicSharedMemorySize,
                            SMEM_BYTES);
  flash_attn<<<kQ / 32, 256, SMEM_BYTES, stream>>>(qk_bf, keys_bf, y_bf, mask, out);
}
